// QLoRALinear_30227979829337
// MI455X (gfx1250) — matrix operand order repaired and checked by reading, not compiled, not run on hardware
//
#include <hip/hip_runtime.h>

#define IN_F    4096
#define OUT_F   4096
#define M_TOT   8192
#define RANK    16
#define SCALING 2.0f

#define BM 128
#define BN 256
#define BK 32
#define LDSS 40                 // LDS row stride in ushorts (80 B = 20 banks), conflict-free frag reads
#define XTILE (BM * LDSS)
#define WTILE (BN * LDSS)
#define BUFSZ (XTILE + WTILE)   // one double-buffer stage (ushorts)

#if __has_builtin(__builtin_amdgcn_global_load_async_to_lds_b128)
#define HAVE_ASYNC 1
#else
#define HAVE_ASYNC 0
#endif

typedef __attribute__((ext_vector_type(16))) __bf16          bf16x16;
typedef __attribute__((ext_vector_type(8)))  float           floatx8;
typedef __attribute__((ext_vector_type(4)))  unsigned int    uintx4;
typedef __attribute__((ext_vector_type(4)))  int             intx4;
typedef __attribute__((ext_vector_type(4)))  unsigned short  ushortx4;
typedef __attribute__((ext_vector_type(8)))  unsigned short  ushortx8;

typedef __attribute__((address_space(1))) intx4 as1_intx4;   // global v4i (builtin arg type)
typedef __attribute__((address_space(3))) intx4 as3_intx4;   // LDS v4i

struct BfPair { uintx4 lo; uintx4 hi; };

__constant__ float NF4_LUT[16] = {
  -1.0f, -0.6962f, -0.5251f, -0.3949f, -0.2844f, -0.1848f, -0.0911f, 0.0f,
   0.0796f, 0.1609f, 0.2461f, 0.3379f, 0.4407f, 0.5626f, 0.723f, 1.0f };

__device__ __forceinline__ unsigned short f2bf(float f) {
  unsigned u = __builtin_bit_cast(unsigned, f);
  u += 0x7FFFu + ((u >> 16) & 1u);              // round-to-nearest-even
  return (unsigned short)(u >> 16);
}

__device__ __forceinline__ bf16x16 make_frag(uintx4 c0, uintx4 c1) {
  BfPair p; p.lo = c0; p.hi = c1;
  return __builtin_bit_cast(bf16x16, p);
}

// async 16B global -> LDS (ASYNCcnt path); falls back to a plain vector copy
__device__ __forceinline__ void copy_b128(const unsigned short* g, unsigned short* l) {
#if HAVE_ASYNC
  __builtin_amdgcn_global_load_async_to_lds_b128((as1_intx4*)g, (as3_intx4*)l, 0, 0);
#else
  *(ushortx8*)l = *(const ushortx8*)g;
#endif
}

__device__ __forceinline__ void wait_async0() {
#if HAVE_ASYNC
 #if __has_builtin(__builtin_amdgcn_s_wait_asynccnt)
  __builtin_amdgcn_s_wait_asynccnt(0);
 #else
  asm volatile("s_wait_asynccnt 0x0" ::: "memory");
 #endif
#endif
}

// ---------------- kernel 1: xa[m,r] = SCALING * sum_k x[m,k] * A[r,k] ----------------
__global__ __launch_bounds__(256) void lora_xa_kernel(
    const float* __restrict__ X, const float* __restrict__ A, float* __restrict__ XA) {
  int gid = blockIdx.x * 256 + threadIdx.x;     // M_TOT*RANK = 131072 threads
  int m = gid >> 4, r = gid & 15;
  const float* xr = X + (size_t)m * IN_F;
  const float* ar = A + (size_t)r * IN_F;
  float acc = 0.f;
  #pragma unroll 4
  for (int k = 0; k < IN_F; k += 4) {
    float4 xv = *(const float4*)(xr + k);
    float4 av = *(const float4*)(ar + k);
    acc = fmaf(xv.x, av.x, acc);
    acc = fmaf(xv.y, av.y, acc);
    acc = fmaf(xv.z, av.z, acc);
    acc = fmaf(xv.w, av.w, acc);
  }
  XA[gid] = acc * SCALING;
}

// ---------------- kernel 2: NF4 -> bf16 W_eff pre-pass ----------------
__global__ __launch_bounds__(256) void dequant_w_kernel(
    const int* __restrict__ PK, const float* __restrict__ SC,
    unsigned short* __restrict__ WB) {
  int g = blockIdx.x * 256 + threadIdx.x;       // int4 = 8 weights
  int4 pv = *(const int4*)(PK + (size_t)g * 4);
  float s = SC[g >> 3];
  ushortx8 o;
  int v;
  v = pv.x; o[0] = f2bf(NF4_LUT[v & 15] * s); o[1] = f2bf(NF4_LUT[(v >> 4) & 15] * s);
  v = pv.y; o[2] = f2bf(NF4_LUT[v & 15] * s); o[3] = f2bf(NF4_LUT[(v >> 4) & 15] * s);
  v = pv.z; o[4] = f2bf(NF4_LUT[v & 15] * s); o[5] = f2bf(NF4_LUT[(v >> 4) & 15] * s);
  v = pv.w; o[6] = f2bf(NF4_LUT[v & 15] * s); o[7] = f2bf(NF4_LUT[(v >> 4) & 15] * s);
  *(ushortx8*)(WB + (size_t)g * 8) = o;
}

// ---------------- kernel 2b: x f32 -> bf16 pre-pass ----------------
__global__ __launch_bounds__(256) void cvt_x_kernel(
    const float* __restrict__ X, unsigned short* __restrict__ XB) {
  int g = blockIdx.x * 256 + threadIdx.x;       // 8 floats per thread
  const float4* xp = (const float4*)(X + (size_t)g * 8);
  float4 a = xp[0], b = xp[1];
  ushortx8 o;
  o[0] = f2bf(a.x); o[1] = f2bf(a.y); o[2] = f2bf(a.z); o[3] = f2bf(a.w);
  o[4] = f2bf(b.x); o[5] = f2bf(b.y); o[6] = f2bf(b.z); o[7] = f2bf(b.w);
  *(ushortx8*)(XB + (size_t)g * 8) = o;
}

// ---------------- kernel 3: pipelined WMMA GEMM, WMMA LoRA epilogue ----------------
// MODE 2: x and W pre-converted to bf16 -> async LDS staging, zero staging VALU
// MODE 1: W pre-converted (async), x converted in-loop
// MODE 0: everything in-loop (small-ws fallback)
template <int MODE>
__global__ __launch_bounds__(256) void qlora_gemm_kernel(
    const float* __restrict__ X, const unsigned short* __restrict__ XB,
    const int* __restrict__ PK, const float* __restrict__ SC,
    const unsigned short* __restrict__ WB,
    const float* __restrict__ LB, const float* __restrict__ BIAS,
    const float* __restrict__ XA, float* __restrict__ OUT)
{
  __shared__ alignas(16) unsigned short lds[2 * BUFSZ];

  const int t    = threadIdx.x;
  const int lane = t & 31;
  const int w    = t >> 5;
  const int wm   = w >> 2;          // 0..1  (64-row wave strip)
  const int wn   = w & 3;           // 0..3  (64-col wave strip)
  const int m0   = blockIdx.y * BM;
  const int n0   = blockIdx.x * BN;
  const int half = lane >> 4;       // 0/1
  const int l16  = lane & 15;

  floatx8 acc[4][4];
  for (int i = 0; i < 4; ++i)
    for (int j = 0; j < 4; ++j)
      #pragma unroll
      for (int e = 0; e < 8; ++e) acc[i][j][e] = 0.0f;

  // ---- staging helpers ----
  auto stage_x_bf = [&](int kc, unsigned short* xd) {         // MODE 2: bf16 x, async
    #pragma unroll
    for (int q = 0; q < 2; ++q) {
      int g = q * 256 + t, r = g >> 2, c = g & 3;             // BM rows x 4 b128 chunks
      copy_b128(XB + (size_t)(m0 + r) * IN_F + kc + c * 8, xd + r * LDSS + c * 8);
    }
  };
  auto stage_w_bf = [&](int kc, unsigned short* wd) {         // MODE 1/2: bf16 W, async
    #pragma unroll
    for (int q = 0; q < 4; ++q) {
      int g = q * 256 + t, r = g >> 2, c = g & 3;             // BN rows x 4 b128 chunks
      copy_b128(WB + (size_t)(n0 + r) * IN_F + kc + c * 8, wd + r * LDSS + c * 8);
    }
  };
  auto fetch_x = [&](int kc, float4 (&px)[4]) {               // MODE 0/1: f32 x
    #pragma unroll
    for (int q = 0; q < 4; ++q) {
      int f = q * 256 + t, r = f >> 3, c4 = f & 7;
      px[q] = *(const float4*)(X + (size_t)(m0 + r) * IN_F + kc + c4 * 4);
    }
  };
  auto store_x = [&](unsigned short* xd, const float4 (&px)[4]) {
    #pragma unroll
    for (int q = 0; q < 4; ++q) {
      int f = q * 256 + t, r = f >> 3, c4 = f & 7;
      ushortx4 hv;
      hv[0] = f2bf(px[q].x); hv[1] = f2bf(px[q].y);
      hv[2] = f2bf(px[q].z); hv[3] = f2bf(px[q].w);
      *(ushortx4*)(xd + r * LDSS + c4 * 4) = hv;
    }
  };
  auto fetch_ws = [&](int kc, int4 (&pi)[4], float (&ps)[4]) { // MODE 0: NF4 inline
    #pragma unroll
    for (int q = 0; q < 4; ++q) {
      int g = q * 256 + t, r = g >> 2, seg = g & 3;
      size_t blk = (size_t)(n0 + r) * (IN_F / 64) + (kc >> 6);
      pi[q] = *(const int4*)(PK + blk * 32 + ((kc & 63) >> 1) + seg * 4);
      ps[q] = SC[blk];
    }
  };
  auto store_ws = [&](unsigned short* wd, const int4 (&pi)[4], const float (&ps)[4]) {
    #pragma unroll
    for (int q = 0; q < 4; ++q) {
      int g = q * 256 + t, r = g >> 2, seg = g & 3;
      float s = ps[q];
      ushortx8 o;
      int v;
      v = pi[q].x; o[0] = f2bf(NF4_LUT[v & 15] * s); o[1] = f2bf(NF4_LUT[(v >> 4) & 15] * s);
      v = pi[q].y; o[2] = f2bf(NF4_LUT[v & 15] * s); o[3] = f2bf(NF4_LUT[(v >> 4) & 15] * s);
      v = pi[q].z; o[4] = f2bf(NF4_LUT[v & 15] * s); o[5] = f2bf(NF4_LUT[(v >> 4) & 15] * s);
      v = pi[q].w; o[6] = f2bf(NF4_LUT[v & 15] * s); o[7] = f2bf(NF4_LUT[(v >> 4) & 15] * s);
      *(ushortx8*)(wd + r * LDSS + seg * 8) = o;
    }
  };

  // one 128x256x32 MMA round from an LDS buffer (contiguous-K frags, same k order for A and B)
  auto mma_step = [&](const unsigned short* xs, const unsigned short* ws) {
    bf16x16 bfrag[4];
    #pragma unroll
    for (int j = 0; j < 4; ++j) {
      int row = wn * 64 + j * 16 + l16;                       // B: lane = N, 16 contiguous K
      const unsigned short* p = ws + row * LDSS + half * 16;
      bfrag[j] = make_frag(*(const uintx4*)(p), *(const uintx4*)(p + 8));
    }
    #pragma unroll
    for (int i = 0; i < 4; ++i) {
      int row = wm * 64 + i * 16 + l16;                       // A: lane = M, 16 contiguous K
      const unsigned short* p = xs + row * LDSS + half * 16;
      bf16x16 af = make_frag(*(const uintx4*)(p), *(const uintx4*)(p + 8));
      #pragma unroll
      for (int j = 0; j < 4; ++j)
        acc[i][j] = __builtin_amdgcn_wmma_f32_16x16x32_bf16(
            false, af, false, bfrag[j], (short)0, acc[i][j], false, false);
    }
  };

  // ---- prologue: stage chunk 0 into buffer 0 ----
  {
    if constexpr (MODE == 2) {
      stage_x_bf(0, lds);
    } else {
      float4 px[4]; fetch_x(0, px); store_x(lds, px);
    }
    if constexpr (MODE >= 1) {
      stage_w_bf(0, lds + XTILE);
    } else {
      int4 pi[4]; float ps[4]; fetch_ws(0, pi, ps); store_ws(lds + XTILE, pi, ps);
    }
    wait_async0();
  }
  __syncthreads();

  // ---- main pipelined K loop: async-issue next -> 16 WMMA -> wait -> 1 barrier ----
  int cur = 0;
  for (int kc = 0; kc < IN_F; kc += BK) {
    const bool nxt = (kc + BK) < IN_F;
    unsigned short* nbuf = lds + (cur ^ 1) * BUFSZ;

    if (nxt) {                                   // async staging lands while WMMA runs
      if constexpr (MODE == 2) stage_x_bf(kc + BK, nbuf);
      if constexpr (MODE >= 1) stage_w_bf(kc + BK, nbuf + XTILE);
    }
    float4 px[4]; int4 pwi[4]; float psc[4];
    if (nxt) {
      if constexpr (MODE <= 1) fetch_x(kc + BK, px);
      if constexpr (MODE == 0) fetch_ws(kc + BK, pwi, psc);
    }

    mma_step(lds + cur * BUFSZ, lds + cur * BUFSZ + XTILE);

    if (nxt) {
      if constexpr (MODE <= 1) store_x(nbuf, px);
      if constexpr (MODE == 0) store_ws(nbuf + XTILE, pwi, psc);
    }
    wait_async0();
    __syncthreads();
    cur ^= 1;
  }

  // ---- LoRA rank-16 update as one extra zero-padded WMMA round ----
  {
    unsigned short* xd = lds;             // all compute done (barrier above), reuse buffer 0
    unsigned short* wd = lds + XTILE;
    const ushortx8 z = {0, 0, 0, 0, 0, 0, 0, 0};
    {   // xa tile: BM rows x (16 real + 16 zero)  -- 256 threads = 128 rows x 2 halves
      int r = t >> 1, h = t & 1;
      const float4* xr = (const float4*)(XA + (size_t)(m0 + r) * RANK + h * 8);
      float4 a0 = xr[0], a1 = xr[1];
      ushortx8 o;
      o[0] = f2bf(a0.x); o[1] = f2bf(a0.y); o[2] = f2bf(a0.z); o[3] = f2bf(a0.w);
      o[4] = f2bf(a1.x); o[5] = f2bf(a1.y); o[6] = f2bf(a1.z); o[7] = f2bf(a1.w);
      *(ushortx8*)(xd + r * LDSS + h * 8) = o;
      *(ushortx8*)(xd + r * LDSS + 16 + h * 8) = z;
    }
    {   // lora_B tile: BN rows x (16 real + 16 zero) -- 1 row per thread
      int r = t;
      const float4* br = (const float4*)(LB + (size_t)(n0 + r) * RANK);
      float4 b0 = br[0], b1 = br[1], b2 = br[2], b3 = br[3];
      ushortx8 o0, o1;
      o0[0] = f2bf(b0.x); o0[1] = f2bf(b0.y); o0[2] = f2bf(b0.z); o0[3] = f2bf(b0.w);
      o0[4] = f2bf(b1.x); o0[5] = f2bf(b1.y); o0[6] = f2bf(b1.z); o0[7] = f2bf(b1.w);
      o1[0] = f2bf(b2.x); o1[1] = f2bf(b2.y); o1[2] = f2bf(b2.z); o1[3] = f2bf(b2.w);
      o1[4] = f2bf(b3.x); o1[5] = f2bf(b3.y); o1[6] = f2bf(b3.z); o1[7] = f2bf(b3.w);
      *(ushortx8*)(wd + r * LDSS)      = o0;
      *(ushortx8*)(wd + r * LDSS + 8)  = o1;
      *(ushortx8*)(wd + r * LDSS + 16) = z;
      *(ushortx8*)(wd + r * LDSS + 24) = z;
    }
    __syncthreads();
    mma_step(xd, wd);
  }

  // ---- epilogue: + bias, store (C/D layout: VGPR e -> M = e + 8*half, N = lane&15) ----
  #pragma unroll
  for (int j = 0; j < 4; ++j) {
    int n = n0 + wn * 64 + j * 16 + l16;
    float bb = BIAS[n];
    #pragma unroll
    for (int i = 0; i < 4; ++i) {
      int mb = m0 + wm * 64 + i * 16 + half * 8;
      #pragma unroll
      for (int e = 0; e < 8; ++e)
        OUT[(size_t)(mb + e) * OUT_F + n] = acc[i][j][e] + bb;
    }
  }
}

extern "C" void kernel_launch(void* const* d_in, const int* in_sizes, int n_in,
                              void* d_out, int out_size, void* d_ws, size_t ws_size,
                              hipStream_t stream) {
  const float* X  = (const float*)d_in[0];
  const int*   PK = (const int*)d_in[1];
  const float* SC = (const float*)d_in[2];
  const float* LA = (const float*)d_in[3];
  const float* LB = (const float*)d_in[4];
  const float* BI = (const float*)d_in[5];
  float* OUT = (float*)d_out;

  const size_t xa_bytes = (size_t)M_TOT * RANK * sizeof(float);             // 512 KiB
  const size_t wb_bytes = (size_t)OUT_F * IN_F * sizeof(unsigned short);    // 32 MiB
  const size_t xb_bytes = (size_t)M_TOT * IN_F * sizeof(unsigned short);    // 64 MiB
  float* XA = (float*)d_ws;
  unsigned short* WB = (unsigned short*)((char*)d_ws + xa_bytes);
  unsigned short* XB = (unsigned short*)((char*)d_ws + xa_bytes + wb_bytes);

  int mode = 0;
  if (ws_size >= xa_bytes + wb_bytes + xb_bytes) mode = 2;
  else if (ws_size >= xa_bytes + wb_bytes)       mode = 1;

  lora_xa_kernel<<<(M_TOT * RANK) / 256, 256, 0, stream>>>(X, LA, XA);

  dim3 grid(OUT_F / BN, M_TOT / BM);   // (16, 64)
  if (mode == 2) {
    dequant_w_kernel<<<(OUT_F * IN_F / 8) / 256, 256, 0, stream>>>(PK, SC, WB);
    cvt_x_kernel<<<((size_t)M_TOT * IN_F / 8) / 256, 256, 0, stream>>>(X, XB);
    qlora_gemm_kernel<2><<<grid, 256, 0, stream>>>(X, XB, PK, SC, WB, LB, BI, XA, OUT);
  } else if (mode == 1) {
    dequant_w_kernel<<<(OUT_F * IN_F / 8) / 256, 256, 0, stream>>>(PK, SC, WB);
    qlora_gemm_kernel<1><<<grid, 256, 0, stream>>>(X, XB, PK, SC, WB, LB, BI, XA, OUT);
  } else {
    qlora_gemm_kernel<0><<<grid, 256, 0, stream>>>(X, XB, PK, SC, WB, LB, BI, XA, OUT);
  }
  (void)in_sizes; (void)n_in; (void)out_size;
}
